// DiffusionModel_47605417509457
// MI455X (gfx1250) — compile-verified, matrix operand order checked by reading, not run on hardware
//
#include <hip/hip_runtime.h>

typedef float     v2f  __attribute__((ext_vector_type(2)));
typedef float     v8f  __attribute__((ext_vector_type(8)));
typedef _Float16  v16h __attribute__((ext_vector_type(16)));

#define N_CELLS 4096
#define M_CH    8
#define S_DIM   16
#define H_DIM   128
#define LOG2E   1.4426950408889634f
#define FOURPI  12.566370614359172f

// ---------------------------------------------------------------------------
// Kernel 1: all-pairs steady-state concentrations.
// Block = 256 threads (8 waves), owns one i-tile of 16 rows.
// dist^2 via V_WMMA_F32_16X16X4_F32:  A=[x,y,z,1], B=[-2x,-2y,-2z,|p|^2], C=|p_i|^2
// Per j-tile per wave: 1 WMMA + 8x(sqrt,rcp) + 64 v_exp_f32 (trans-pipe bound,
// VALU dual-issues against it). Next tile prefetched via global_prefetch_b8.
// ---------------------------------------------------------------------------
__global__ __launch_bounds__(256) void conc_kernel(
    const float* __restrict__ pos,        // [N,3]
    const float* __restrict__ radius,     // [N]
    const float* __restrict__ secretion,  // [N,M]
    const int*   __restrict__ parent,     // [N]
    const float* __restrict__ Dc,         // [M]
    const float* __restrict__ Kc,         // [M]
    float*       __restrict__ conc)       // [N,M]  (workspace)
{
    __shared__ float lds_w[8][16][M_CH];  // per-wave partial sums, 4KB

    const int tid  = threadIdx.x;
    const int lane = tid & 31;
    const int wave = tid >> 5;            // 0..7
    const int half = lane >> 4;           // 0/1
    const int l    = lane & 15;           // column / row-within-half index
    const int ibase = blockIdx.x * 16;

    // per-channel constants (tiny, L2/K$-resident)
    float lam[M_CH], coef[M_CH];
#pragma unroll
    for (int m = 0; m < M_CH; ++m) {
        float d = Dc[m], k = Kc[m];
        lam[m]  = -__builtin_amdgcn_sqrtf(k / d) * LOG2E;  // exp(-lam*r) = exp2(lam[m]*r)
        coef[m] = 1.0f / (FOURPI * d);
    }

    // A matrix (16x4 f32): lanes 0-15 hold K={0,1}, lanes 16-31 hold K={2,3}
    const int ia = ibase + l;
    float ax = pos[ia*3+0], ay = pos[ia*3+1], az = pos[ia*3+2];
    v2f A;
    A.x = half ? az   : ax;   // K=0 | K=2
    A.y = half ? 1.0f : ay;   // K=1 | K=3

    // C matrix: C[i][j] = |p_i|^2  (row-constant; lane's rows are v + 8*half)
    v8f Cn;
#pragma unroll
    for (int v = 0; v < 8; ++v) {
        int ic = ibase + v + 8*half;
        float cx = pos[ic*3+0], cy = pos[ic*3+1], cz = pos[ic*3+2];
        Cn[v] = cx*cx + cy*cy + cz*cz;
    }

    float acc[8][M_CH];
#pragma unroll
    for (int v = 0; v < 8; ++v)
#pragma unroll
        for (int m = 0; m < M_CH; ++m) acc[v][m] = 0.0f;

    const int njt = N_CELLS / 16;          // 256 j-tiles, 32 per wave
    for (int jt = wave; jt < njt; jt += 8) {
        const int j = jt*16 + l;

        // prefetch next j-tile for this wave (global_prefetch_b8; no LOADcnt)
        if (jt + 8 < njt) {
            const int jp = (jt + 8)*16 + l;
            __builtin_prefetch(&pos[jp*3], 0, 0);
            __builtin_prefetch(&secretion[jp*M_CH], 0, 0);
            __builtin_prefetch(&radius[jp], 0, 0);
        }

        float bx = pos[j*3+0], by = pos[j*3+1], bz = pos[j*3+2];
        float nrmj = bx*bx + by*by + bz*bz;
        v2f B;
        B.x = half ? (-2.0f*bz) : (-2.0f*bx);   // K=0 | K=2
        B.y = half ? nrmj       : (-2.0f*by);   // K=1 | K=3

        const float rj   = radius[j];
        const float actj = (parent[j] >= 0) ? 1.0f : 0.0f;
        float w[M_CH];
#pragma unroll
        for (int m = 0; m < M_CH; ++m)
            w[m] = secretion[j*M_CH + m] * actj * coef[m];

        // dist^2 tile: D = A x B + C
        v8f d2 = __builtin_amdgcn_wmma_f32_16x16x4_f32(
            false, A, false, B, (short)0, Cn, false, false);

#pragma unroll
        for (int v = 0; v < 8; ++v) {
            float dsq  = d2[v];
            dsq        = dsq > 0.0f ? dsq : 0.0f;       // guard FP cancellation
            float dist = __builtin_amdgcn_sqrtf(dsq + 1e-12f);
            dist       = fmaxf(dist, rj);
            float invd = __builtin_amdgcn_rcpf(dist);
#pragma unroll
            for (int m = 0; m < M_CH; ++m) {
                float e = __builtin_amdgcn_exp2f(dist * lam[m]);  // exp(-lam*dist)
                acc[v][m] = fmaf(e * invd, w[m], acc[v][m]);
            }
        }
    }

    // reduce over the 16 columns (j within tile) inside each lane-half
#pragma unroll
    for (int v = 0; v < 8; ++v) {
#pragma unroll
        for (int m = 0; m < M_CH; ++m) {
            float s = acc[v][m];
            s += __shfl_xor(s, 1, 32);
            s += __shfl_xor(s, 2, 32);
            s += __shfl_xor(s, 4, 32);
            s += __shfl_xor(s, 8, 32);
            if (l == 0) lds_w[wave][v + 8*half][m] = s;   // lanes 0 and 16 write
        }
    }
    __syncthreads();

    // cross-wave reduction + active-target mask + store
    if (tid < 16 * M_CH) {
        int r = tid >> 3, m = tid & 7;
        float s = 0.0f;
#pragma unroll
        for (int wv = 0; wv < 8; ++wv) s += lds_w[wv][r][m];
        int i = ibase + r;
        float acti = (parent[i] >= 0) ? 1.0f : 0.0f;
        conc[i*M_CH + m] = s * acti;
    }
}

// ---------------------------------------------------------------------------
// Kernel 2: MLP  out = state + W2 @ relu(LN(W1 @ [state,conc] + b1)) + b2
// Block = 128 threads (4 waves), owns 16 rows. f16 WMMA, K padded 24->32.
// ---------------------------------------------------------------------------
__device__ __forceinline__ float get_x(const float* state, const float* conc,
                                       int i, int k) {
    if (k < S_DIM)          return state[i*S_DIM + k];
    if (k < S_DIM + M_CH)   return conc[i*M_CH + (k - S_DIM)];
    return 0.0f;
}

__global__ __launch_bounds__(128) void mlp_kernel(
    const float* __restrict__ state,   // [N,16]
    const float* __restrict__ conc,    // [N,8]
    const float* __restrict__ W1,      // [128,24]
    const float* __restrict__ b1,      // [128]
    const float* __restrict__ ln_g,    // [128]
    const float* __restrict__ ln_b,    // [128]
    const float* __restrict__ W2,      // [16,128]
    const float* __restrict__ b2,      // [16]
    float*       __restrict__ out)     // [N,16]
{
    __shared__ float hbuf[16 * H_DIM];       // 8KB  hidden activations
    __shared__ float obuf[4][16][S_DIM];     // 4KB  K-chunk partials
    __shared__ float mu_s[16], sc_s[16];

    const int tid  = threadIdx.x;
    const int lane = tid & 31;
    const int wave = tid >> 5;               // 0..3
    const int half = lane >> 4;
    const int l    = lane & 15;
    const int ibase = blockIdx.x * 16;
    const int i     = ibase + l;

    // ---- layer 1: A = x[i][k] in f16, 16-bit A layout:
    // VGPR p (p<4): lanes<16 K=2p,2p+1 ; lanes>=16 K=8+2p,8+2p+1 ; VGPR 4..7: +16
    v16h Ax;
#pragma unroll
    for (int p = 0; p < 4; ++p) {
        int k0 = 2*p + 8*half;
        Ax[2*p]     = (_Float16)get_x(state, conc, i, k0);
        Ax[2*p + 1] = (_Float16)get_x(state, conc, i, k0 + 1);
        int k1 = k0 + 16;
        Ax[8 + 2*p]     = (_Float16)get_x(state, conc, i, k1);
        Ax[8 + 2*p + 1] = (_Float16)get_x(state, conc, i, k1 + 1);
    }

    // each wave computes 2 of the 8 column tiles of H=128
#pragma unroll
    for (int t = 0; t < 2; ++t) {
        const int n = (wave*2 + t)*16 + l;   // hidden unit index (B column)
        v16h Bw;
#pragma unroll
        for (int p = 0; p < 4; ++p) {
            int k0 = 2*p + 8*half;
            Bw[2*p]     = (_Float16)(k0     < 24 ? W1[n*24 + k0]     : 0.0f);
            Bw[2*p + 1] = (_Float16)(k0 + 1 < 24 ? W1[n*24 + k0 + 1] : 0.0f);
            int k1 = k0 + 16;
            Bw[8 + 2*p]     = (_Float16)(k1     < 24 ? W1[n*24 + k1]     : 0.0f);
            Bw[8 + 2*p + 1] = (_Float16)(k1 + 1 < 24 ? W1[n*24 + k1 + 1] : 0.0f);
        }
        v8f c;
        float bias = b1[n];
#pragma unroll
        for (int v = 0; v < 8; ++v) c[v] = bias;

        v8f h = __builtin_amdgcn_wmma_f32_16x16x32_f16(
            false, Ax, false, Bw, (short)0, c, false, false);

#pragma unroll
        for (int v = 0; v < 8; ++v)
            hbuf[(v + 8*half)*H_DIM + n] = h[v];
    }
    __syncthreads();

    // ---- LayerNorm stats: one thread per row
    if (tid < 16) {
        float s = 0.0f, sq = 0.0f;
        for (int kk = 0; kk < H_DIM; ++kk) {
            float v = hbuf[tid*H_DIM + kk];
            s += v; sq += v*v;
        }
        float mu  = s * (1.0f / H_DIM);
        float var = sq * (1.0f / H_DIM) - mu*mu;
        mu_s[tid] = mu;
        sc_s[tid] = __builtin_amdgcn_rsqf(var + 1e-5f);
    }
    __syncthreads();

    // ---- normalize + affine + ReLU (thread = one hidden column)
    {
        const float g = ln_g[tid], bb = ln_b[tid];
#pragma unroll
        for (int r = 0; r < 16; ++r) {
            float v = hbuf[r*H_DIM + tid];
            v = (v - mu_s[r]) * sc_s[r] * g + bb;
            hbuf[r*H_DIM + tid] = v > 0.0f ? v : 0.0f;
        }
    }
    __syncthreads();

    // ---- layer 2: each wave handles K-chunk [wave*32, wave*32+32)
    {
        v16h Ah, Bh;
#pragma unroll
        for (int p = 0; p < 4; ++p) {
            int k0 = wave*32 + 2*p + 8*half;
            Ah[2*p]     = (_Float16)hbuf[l*H_DIM + k0];
            Ah[2*p + 1] = (_Float16)hbuf[l*H_DIM + k0 + 1];
            Bh[2*p]     = (_Float16)W2[l*H_DIM + k0];
            Bh[2*p + 1] = (_Float16)W2[l*H_DIM + k0 + 1];
            int k1 = k0 + 16;
            Ah[8 + 2*p]     = (_Float16)hbuf[l*H_DIM + k1];
            Ah[8 + 2*p + 1] = (_Float16)hbuf[l*H_DIM + k1 + 1];
            Bh[8 + 2*p]     = (_Float16)W2[l*H_DIM + k1];
            Bh[8 + 2*p + 1] = (_Float16)W2[l*H_DIM + k1 + 1];
        }
        v8f co;
        float bias = (wave == 0) ? b2[l] : 0.0f;   // add bias exactly once
#pragma unroll
        for (int v = 0; v < 8; ++v) co[v] = bias;

        v8f o = __builtin_amdgcn_wmma_f32_16x16x32_f16(
            false, Ah, false, Bh, (short)0, co, false, false);

#pragma unroll
        for (int v = 0; v < 8; ++v)
            obuf[wave][v + 8*half][l] = o[v];
    }
    __syncthreads();

    // ---- reduce K-chunks, residual add, store (256 outputs / 128 threads)
    for (int e = tid; e < 16*S_DIM; e += 128) {
        int r = e >> 4, cidx = e & 15;
        float s = obuf[0][r][cidx] + obuf[1][r][cidx]
                + obuf[2][r][cidx] + obuf[3][r][cidx];
        out[(ibase + r)*S_DIM + cidx] = state[(ibase + r)*S_DIM + cidx] + s;
    }
}

// ---------------------------------------------------------------------------
extern "C" void kernel_launch(void* const* d_in, const int* in_sizes, int n_in,
                              void* d_out, int out_size, void* d_ws, size_t ws_size,
                              hipStream_t stream) {
    const float* position  = (const float*)d_in[0];
    const float* radius    = (const float*)d_in[1];
    const float* secretion = (const float*)d_in[2];
    const float* state     = (const float*)d_in[3];
    const int*   parent    = (const int*)  d_in[4];
    const float* Dc        = (const float*)d_in[5];
    const float* Kc        = (const float*)d_in[6];
    const float* W1        = (const float*)d_in[7];
    const float* b1        = (const float*)d_in[8];
    const float* ln_g      = (const float*)d_in[9];
    const float* ln_b      = (const float*)d_in[10];
    const float* W2        = (const float*)d_in[11];
    const float* b2        = (const float*)d_in[12];
    float* out  = (float*)d_out;
    float* conc = (float*)d_ws;               // [N, M] scratch, fully overwritten

    conc_kernel<<<N_CELLS/16, 256, 0, stream>>>(position, radius, secretion,
                                                parent, Dc, Kc, conc);
    mlp_kernel<<<N_CELLS/16, 128, 0, stream>>>(state, conc, W1, b1, ln_g, ln_b,
                                               W2, b2, out);
}
